// CS_MSA_7275674600384
// MI455X (gfx1250) — matrix operand order repaired and checked by reading, not compiled, not run on hardware
//
#include <hip/hip_runtime.h>

// ---------------------------------------------------------------------------
// Dual-stream 8x8 window attention (Swin-like) for MI455X / gfx1250.
// One workgroup (256 threads = 8 wave32) per window; all GEMMs via
// v_wmma_f32_16x16x32_bf16.
// Per-launch pre-pack into d_ws:
//   - weights -> fragment-major bf16 (B-frag load = 2x global_load_b128)
//   - pos_emb -> C-tile-fragment-major fp32 (tile add = 2x global_load_b128)
// V/PV stored transposed in LDS so attn@V B-fragments are contiguous 32B loads.
// ---------------------------------------------------------------------------

#define DIM    224
#define HEADS  8
#define DHEAD  28
#define INNER  224
#define SEQ    64
#define HH     256
#define WW     256

static constexpr float SCALE = 0.18898223650461363f; // 28^-0.5

typedef __attribute__((ext_vector_type(16))) __bf16 v16bf;
typedef __attribute__((ext_vector_type(8)))  float  v8f;

union Frag { v16bf v; unsigned u[8]; uint2 u2[4]; uint4 u4[2]; };

// ---------------------------------------------------------------------------
// Weight pre-pack: fragment-major bf16, contiguous k order (k = 16*(lane/16) + e).
// ---------------------------------------------------------------------------
__global__ void pack_weights_kernel(const float* __restrict__ W,
                                    __bf16* __restrict__ dst,
                                    int Kdim, int Ndim) {
  int idx = blockIdx.x * blockDim.x + threadIdx.x; // one per (frag,lane)
  int kT = Kdim >> 5;
  int nT = Ndim >> 4;
  int total = kT * nT * 32;
  if (idx >= total) return;
  int lane = idx & 31;
  int frag = idx >> 5;
  int kt = frag % kT;
  int nt = frag / kT;
  int n  = (nt << 4) + (lane & 15);
  int kb = (kt << 5) + ((lane >> 4) << 4);
  __bf16* d = dst + (size_t)idx * 16;
#pragma unroll
  for (int e = 0; e < 16; ++e)
    d[e] = (__bf16)W[(size_t)(kb + e) * Ndim + n];
}

// ---------------------------------------------------------------------------
// pos_emb pre-pack: C-tile-fragment-major fp32.
// Tile (pair, it, jt): lane holds rows rb..rb+7 (rb = (lane>>4)*8 + it*16) of
// column jt*16 + (lane&15) -> 8 consecutive floats per lane.
// ---------------------------------------------------------------------------
__global__ void pack_pos_kernel(const float* __restrict__ pos1,
                                const float* __restrict__ pos2,
                                float* __restrict__ dst) {
  int idx = blockIdx.x * blockDim.x + threadIdx.x; // one per (tile,lane)
  const int total = 16 * 4 * 4 * 32;               // pairs * it * jt * lanes
  if (idx >= total) return;
  int lane = idx & 31;
  int tile = idx >> 5;
  int jt   = tile & 3;
  int it   = (tile >> 2) & 3;
  int pair = tile >> 4;                 // stream*8 + h
  const float* P = ((pair >> 3) ? pos2 : pos1) + (pair & 7) * SEQ * SEQ;
  int col = (jt << 4) + (lane & 15);
  int rb  = (it << 4) + ((lane >> 4) << 3);
  float* d = dst + (size_t)idx * 8;
#pragma unroll
  for (int i = 0; i < 8; ++i)
    d[i] = P[(rb + i) * SEQ + col];
}

// B fragment from packed weights: 2x global_load_b128.
__device__ __forceinline__ Frag load_b_packed(const __bf16* Wp, int kT, int kt, int nt) {
  const uint4* p = (const uint4*)(Wp + ((size_t)(nt * kT + kt) * 32 + (threadIdx.x & 31)) * 16);
  Frag f;
  f.u4[0] = p[0];
  f.u4[1] = p[1];
  return f;
}

// A fragment (16x32, MxK) from row-major bf16 LDS, no guards (pad handled on B).
__device__ __forceinline__ Frag load_a(const __bf16* A, int lda, int row0, int k0) {
  int lane = threadIdx.x & 31;
  const uint2* p = (const uint2*)(A + (row0 + (lane & 15)) * lda + k0 + ((lane >> 4) << 4));
  Frag f;
  f.u2[0] = p[0];  // k +0..3
  f.u2[1] = p[1];  // k +4..7
  f.u2[2] = p[2];  // k +8..11
  f.u2[3] = p[3];  // k +12..15
  return f;
}

// B fragment for sim = q @ k^T : B[d][j] = Klds[j][col0+d]; d contiguous.
// Only d=0..27 valid -> zero the hi-half's last two pairs.
__device__ __forceinline__ Frag load_b_kT(const __bf16* K, int ldk, int j0, int col0) {
  int lane = threadIdx.x & 31;
  bool hi = lane >= 16;
  const uint2* p = (const uint2*)(K + (j0 + (lane & 15)) * ldk + col0 + (hi ? 16 : 0));
  Frag f;
  f.u2[0] = p[0];
  f.u2[1] = p[1];
  f.u2[2] = p[2];
  uint2 t = p[3];
  if (hi) { t.x = 0u; t.y = 0u; }   // abs d 28..31 -> zero pad
  f.u2[3] = t;
  return f;
}

// B fragment for out = attn @ V from transposed vT[d][seq]: contiguous in K.
__device__ __forceinline__ Frag load_b_vt(const __bf16* VT, int k0, int col0) {
  int lane = threadIdx.x & 31;
  const uint4* p = (const uint4*)(VT + (col0 + (lane & 15)) * SEQ + k0 + ((lane >> 4) << 4));
  Frag f;
  f.u4[0] = p[0];
  f.u4[1] = p[1];
  return f;
}

__device__ __forceinline__ v8f wmma_bf16(Frag a, Frag b, v8f c) {
  return __builtin_amdgcn_wmma_f32_16x16x32_bf16(false, a.v, false, b.v,
                                                 (short)0, c, false, false);
}

__global__ __launch_bounds__(256) void winattn_kernel(
    const float* __restrict__ x,    const float* __restrict__ px,
    const float* __restrict__ posP,
    const __bf16* __restrict__ Wq,  const __bf16* __restrict__ Wkv,
    const __bf16* __restrict__ Wkv2,const __bf16* __restrict__ Wvf,
    const float* __restrict__ bvf,  const __bf16* __restrict__ Wout,
    const float* __restrict__ bout, float* __restrict__ out)
{
  // CDNA5: up to 320KB LDS per workgroup. Total here: 240KB.
  __shared__ __bf16 s_fused[SEQ * 2 * INNER]; // 57344B; head doubles as staging
  __shared__ __bf16 s_q  [SEQ * INNER];
  __shared__ __bf16 s_k  [SEQ * INNER];
  __shared__ __bf16 s_vt [INNER * SEQ];       // transposed: [d][seq]
  __shared__ __bf16 s_pk [SEQ * INNER];
  __shared__ __bf16 s_pvt[INNER * SEQ];
  __shared__ __bf16 s_scr[8 * 16 * SEQ];      // per-wave attn re-layout bounce
  __shared__ __bf16 s_mid[SEQ * INNER];

  const int w   = blockIdx.x;
  const int bi  = w >> 10;
  const int rem = w & 1023;
  const int hy  = rem >> 5;
  const int hx  = rem & 31;
  const long base = (((long)bi * HH + hy * 8) * WW + hx * 8) * DIM;

  const int tid  = threadIdx.x;
  const int wave = tid >> 5;
  const int lane = tid & 31;

  // ---- stage one window (64 rows x 224 fp32 -> bf16 LDS) -------------------
  auto stage = [&](const float* __restrict__ src, __bf16* dst) {
    int s    = tid >> 2;
    int part = tid & 3;
    const float4* p = (const float4*)(src + base +
        (((s >> 3) * WW) + (s & 7)) * DIM + part * 56);
    __bf16* d = dst + s * DIM + part * 56;
#pragma unroll
    for (int r4 = 0; r4 < 14; ++r4) {
      float4 f = p[r4];
      d[r4 * 4 + 0] = (__bf16)f.x;
      d[r4 * 4 + 1] = (__bf16)f.y;
      d[r4 * 4 + 2] = (__bf16)f.z;
      d[r4 * 4 + 3] = (__bf16)f.w;
    }
  };

  // ---- generic [64 x K] @ [K x N], result bf16 LDS -------------------------
  auto gemm = [&](const __bf16* A, int lda, const __bf16* Wp, int Kdim, int Ndim,
                  __bf16* D0, __bf16* D1, int splitN, bool vtD1,
                  float scale, const float* bias) {
    int nT = Ndim >> 4;
    int kT = Kdim >> 5;
    for (int nt = wave; nt < nT; nt += 8) {
      int n0 = nt << 4;
      v8f acc[4] = {};
      for (int kt = 0; kt < kT; ++kt) {
        Frag bf = load_b_packed(Wp, kT, kt, nt);
#pragma unroll
        for (int it = 0; it < 4; ++it) {
          Frag af = load_a(A, lda, it << 4, kt << 5);
          acc[it] = wmma_bf16(af, bf, acc[it]);
        }
      }
      bool second = n0 >= splitN;
      int c = (second ? n0 - splitN : n0) + (lane & 15);
      float bs = bias ? bias[n0 + (lane & 15)] : 0.f;
      if (second && vtD1) {
#pragma unroll
        for (int it = 0; it < 4; ++it) {
          int mb = (it << 4) + ((lane >> 4) << 3);
#pragma unroll
          for (int i = 0; i < 8; ++i)
            D1[c * SEQ + mb + i] = (__bf16)(acc[it][i] * scale + bs);
        }
      } else {
        __bf16* D = second ? D1 : D0;
#pragma unroll
        for (int it = 0; it < 4; ++it) {
          int mb = (it << 4) + ((lane >> 4) << 3);
#pragma unroll
          for (int i = 0; i < 8; ++i)
            D[(mb + i) * INNER + c] = (__bf16)(acc[it][i] * scale + bs);
        }
      }
    }
  };

  // ===== phase 1: x -> q,k,vT ==============================================
  stage(x, s_fused);
  __syncthreads();
  gemm(s_fused, DIM, Wq,  DIM,     INNER, s_q, s_q,  INNER, false, SCALE, nullptr);
  gemm(s_fused, DIM, Wkv, DIM, 2 * INNER, s_k, s_vt, INNER, true,  1.0f,  nullptr);
  __syncthreads();

  // ===== phase 2: px -> pk,pvT =============================================
  stage(px, s_fused);
  __syncthreads();
  gemm(s_fused, DIM, Wkv2, DIM, 2 * INNER, s_pk, s_pvt, INNER, true, 1.0f, nullptr);
  __syncthreads();

  // ===== phase 3: attention (8 waves x 16 head/stream pairs) ===============
  __bf16* ascr = s_scr + wave * (16 * SEQ);
  for (int pair = wave; pair < 16; pair += 8) {
    int stream = pair >> 3;
    int h      = pair & 7;
    const __bf16* KB = stream ? s_pk  : s_k;
    const __bf16* VT = stream ? s_pvt : s_vt;
    int hcol = h * DHEAD;

    for (int it = 0; it < 4; ++it) {
      // sim tile-row [16 x 64], single K-step (28 padded to 32 on B side)
      v8f acc[4] = {};
      Frag aq = load_a(s_q, DIM, it << 4, hcol);
#pragma unroll
      for (int jt = 0; jt < 4; ++jt) {
        Frag bk = load_b_kT(KB, DIM, jt << 4, hcol);
        acc[jt] = wmma_bf16(aq, bk, acc[jt]);
      }

      // + pos_emb (packed C-tile-fragment-major: 2x b128 per jt)
#pragma unroll
      for (int jt = 0; jt < 4; ++jt) {
        const float4* pp = (const float4*)(posP +
            ((size_t)(((pair << 2) + it) << 2 | jt) * 32 + lane) * 8);
        float4 p0 = pp[0], p1 = pp[1];
        acc[jt][0] += p0.x; acc[jt][1] += p0.y;
        acc[jt][2] += p0.z; acc[jt][3] += p0.w;
        acc[jt][4] += p1.x; acc[jt][5] += p1.y;
        acc[jt][6] += p1.z; acc[jt][7] += p1.w;
      }

      // register softmax: each row lives across 16 lanes of one half-wave
      float inv[8];
#pragma unroll
      for (int i = 0; i < 8; ++i) {
        float m = fmaxf(fmaxf(acc[0][i], acc[1][i]), fmaxf(acc[2][i], acc[3][i]));
#pragma unroll
        for (int d = 1; d < 16; d <<= 1) m = fmaxf(m, __shfl_xor(m, d, 32));
        float s = 0.f;
#pragma unroll
        for (int jt = 0; jt < 4; ++jt) {
          float e = __expf(acc[jt][i] - m);
          acc[jt][i] = e;
          s += e;
        }
#pragma unroll
        for (int d = 1; d < 16; d <<= 1) s += __shfl_xor(s, d, 32);
        inv[i] = 1.f / s;
      }

      // bounce attn through wave-local LDS: C-tile layout -> A-fragment layout
      {
        int col = lane & 15;
        int rb  = (lane >> 4) << 3;
#pragma unroll
        for (int jt = 0; jt < 4; ++jt)
#pragma unroll
          for (int i = 0; i < 8; ++i)
            ascr[(rb + i) * SEQ + (jt << 4) + col] = (__bf16)(acc[jt][i] * inv[i]);
      }

      // out tile [16 x 28] = attn[16 x 64] @ V_h[64 x 28]
      v8f oc[2] = {};
#pragma unroll
      for (int kt = 0; kt < 2; ++kt) {
        Frag aa = load_a(ascr, SEQ, 0, kt << 5);
#pragma unroll
        for (int nt = 0; nt < 2; ++nt) {
          Frag bv = load_b_vt(VT, kt << 5, hcol + (nt << 4));
          oc[nt] = wmma_bf16(aa, bv, oc[nt]);
        }
      }

      // scatter into fused [64 x 448] (cols >= 28 of tile 1 dropped)
      int cb  = stream * INNER + hcol;
      int nl  = lane & 15;
      int rb2 = (it << 4) + ((lane >> 4) << 3);
#pragma unroll
      for (int nt = 0; nt < 2; ++nt) {
        int c = (nt << 4) + nl;
        if (c < DHEAD)
#pragma unroll
          for (int i = 0; i < 8; ++i)
            s_fused[(rb2 + i) * (2 * INNER) + cb + c] = (__bf16)oc[nt][i];
      }
    }
  }
  __syncthreads();

  // ===== phase 4: fused @ Wvf + bvf -> mid =================================
  gemm(s_fused, 2 * INNER, Wvf, 2 * INNER, INNER, s_mid, s_mid, INNER, false, 1.0f, bvf);
  __syncthreads();

  // ===== phase 5: mid @ Wout + bout -> global (f32) ========================
  for (int nt = wave; nt < 14; nt += 8) {
    int n0 = nt << 4;
    v8f acc[4] = {};
    for (int kt = 0; kt < 7; ++kt) {
      Frag bf = load_b_packed(Wout, 7, kt, nt);
#pragma unroll
      for (int it = 0; it < 4; ++it) {
        Frag af = load_a(s_mid, DIM, it << 4, kt << 5);
        acc[it] = wmma_bf16(af, bf, acc[it]);
      }
    }
    int n = n0 + (lane & 15);
    float bo = bout[n];
#pragma unroll
    for (int it = 0; it < 4; ++it) {
      int mb = (it << 4) + ((lane >> 4) << 3);
#pragma unroll
      for (int i = 0; i < 8; ++i) {
        int s = mb + i;
        out[base + (((s >> 3) * WW) + (s & 7)) * DIM + n] = acc[it][i] + bo;
      }
    }
  }
}

extern "C" void kernel_launch(void* const* d_in, const int* in_sizes, int n_in,
                              void* d_out, int out_size, void* d_ws, size_t ws_size,
                              hipStream_t stream) {
  const float* x    = (const float*)d_in[0];
  const float* px   = (const float*)d_in[1];
  const float* pos1 = (const float*)d_in[2];
  const float* pos2 = (const float*)d_in[3];
  const float* Wq   = (const float*)d_in[4];
  const float* Wkv  = (const float*)d_in[5];
  const float* Wkv2 = (const float*)d_in[6];
  const float* Wvf  = (const float*)d_in[7];
  const float* bvf  = (const float*)d_in[8];
  const float* Wout = (const float*)d_in[9];
  const float* bout = (const float*)d_in[10];

  // d_ws layout: packed bf16 weights (784KB) then packed fp32 pos (256KB)
  __bf16* ws = (__bf16*)d_ws;
  const size_t fQ   = 7  * 14 * 512;   // 50176
  const size_t fKV  = 7  * 28 * 512;   // 100352
  const size_t fVF  = 14 * 14 * 512;   // 100352
  __bf16* pWq   = ws;
  __bf16* pWkv  = pWq   + fQ;
  __bf16* pWkv2 = pWkv  + fKV;
  __bf16* pWvf  = pWkv2 + fKV;
  __bf16* pWout = pWvf  + fVF;
  float*  pPos  = (float*)(pWout + fQ);  // 802816B offset, 16B aligned

  auto pack = [&](const float* W, __bf16* dst, int K, int N) {
    int total = (K >> 5) * (N >> 4) * 32;
    pack_weights_kernel<<<dim3((total + 255) / 256), dim3(256), 0, stream>>>(W, dst, K, N);
  };
  pack(Wq,   pWq,       DIM,     INNER);
  pack(Wkv,  pWkv,      DIM, 2 * INNER);
  pack(Wkv2, pWkv2,     DIM, 2 * INNER);
  pack(Wvf,  pWvf,  2 * INNER,   INNER);
  pack(Wout, pWout,     DIM,     INNER);
  pack_pos_kernel<<<dim3(8192 / 256), dim3(256), 0, stream>>>(pos1, pos2, pPos);

  const int n_windows = 2 * (HH / 8) * (WW / 8); // 2048
  winattn_kernel<<<dim3(n_windows), dim3(256), 0, stream>>>(
      x, px, pPos, pWq, pWkv, pWkv2, pWvf, bvf, pWout, bout, (float*)d_out);
}
